// SGCNN_55233279427067
// MI455X (gfx1250) — matrix operand order repaired and checked by reading, not compiled, not run on hardware
//
#include <hip/hip_runtime.h>

typedef __attribute__((ext_vector_type(16))) _Float16 v16h;
typedef __attribute__((ext_vector_type(8)))  _Float16 v8h;
typedef __attribute__((ext_vector_type(4)))  _Float16 v4h;
typedef __attribute__((ext_vector_type(8)))  float    v8f;
typedef __attribute__((ext_vector_type(4)))  float    v4f;

#define DEV __device__ __forceinline__

namespace sg {
constexpr int NSn = 30000, ESn = 360000, NBn = 30000, EBn = 360000;
constexpr int NF = 92, EF = 41, D = 64, LY = 3, FCn = 128, NGn = 64;
constexpr int KC  = 2 * D + EF;   // 169
constexpr int KCP = 192;          // padded to 6*32
constexpr int KS_E = 6, NT_E = 8; // edge gemm: 6 k-steps, 8 n-tiles (gate 0..3, scr 4..7)
constexpr int KEP = 96;           // emb K padded to 3*32
constexpr int KS_N = 3, NT_N = 4;
constexpr float BNEPS = 1e-5f;
constexpr int EMB_H  = KS_N * NT_N * 32 * 16;   // 6144 halves
constexpr int CONV_H = KS_E * NT_E * 32 * 16;   // 24576 halves / layer
}

DEV float sgm(float x)   { return 1.f / (1.f + __expf(-x)); }
DEV float splus(float x) { return fmaxf(x, 0.f) + log1pf(__expf(-fabsf(x))); }

DEV v8f wmma16(v16h a, v16h b, v8f c) {
    return __builtin_amdgcn_wmma_f32_16x16x32_f16(false, a, false, b, (short)0, c, false, false);
}

// ---- pre-swizzle all weights into WMMA B-fragment layout (fp16, K padded) ----
__global__ void k_prep(const float* __restrict__ embS, const float* __restrict__ embB,
                       const float* __restrict__ convS, const float* __restrict__ convB,
                       _Float16* __restrict__ w16) {
    using namespace sg;
    int tid = blockIdx.x * blockDim.x + threadIdx.x;
    const int EMB_TOT  = 2 * EMB_H;         // 12288
    const int CONV_TOT = 2 * LY * CONV_H;   // 147456
    if (tid < EMB_TOT) {
        int br = tid / EMB_H, r = tid % EMB_H;
        int j = r & 15, lane = (r >> 4) & 31, nt = (r >> 9) & 3, ks = r >> 11;
        int k = ks * 32 + ((lane < 16) ? 0 : 8) + (j & 7) + ((j & 8) ? 16 : 0);   // B-frag: lanes 0-15 K {0..7,16..23}, 16-31 K {8..15,24..31}
        int col = nt * 16 + (lane & 15);
        const float* W = br ? embB : embS;
        w16[tid] = (_Float16)((k < NF) ? W[k * D + col] : 0.f);
    } else if (tid < EMB_TOT + CONV_TOT) {
        int t = tid - EMB_TOT;
        int br = t / (LY * CONV_H), r0 = t % (LY * CONV_H);
        int layer = r0 / CONV_H, r = r0 % CONV_H;
        int j = r & 15, lane = (r >> 4) & 31, nt = (r >> 9) & 7, ks = r >> 12;
        int k = ks * 32 + ((lane < 16) ? 0 : 8) + (j & 7) + ((j & 8) ? 16 : 0);
        int path = nt >> 2;
        int col = (nt & 3) * 16 + (lane & 15);
        const float* W = br ? convB : convS;
        float v = (k < KC) ? W[((layer * 2 + path) * KC + k) * D + col] : 0.f;
        w16[tid] = (_Float16)v;
    }
}

__global__ void k_fill(float* __restrict__ p, int n) {
    int i = blockIdx.x * blockDim.x + threadIdx.x;
    if (i < n) p[i] = 0.f;
}

// ---- embedding GEMM (WMMA) + column stats; stores z into hout ----
__global__ void k_emb(const float* __restrict__ x, int n, const _Float16* __restrict__ w16,
                      float* __restrict__ hout, float* __restrict__ stats) {
    using namespace sg;
    __shared__ _Float16 smA[4][16][KEP];
    __shared__ float smS[2 * D];
    int t = threadIdx.x, w = t >> 5, lane = t & 31;
    int tile = blockIdx.x * 4 + w;
    int row = lane & 15;
    int node = tile * 16 + row;
    bool valid = node < n;
    int seg = lane >> 4;
    v4h z4 = {};
    const v4f* xr = (const v4f*)(x + (size_t)node * NF);   // rows are 368B -> 16B aligned
    if (seg == 0) {          // cols 0..47
        if (valid) {
#pragma unroll
            for (int q = 0; q < 12; ++q)
                *(v4h*)&smA[w][row][q * 4] = __builtin_convertvector(xr[q], v4h);
        } else {
#pragma unroll
            for (int q = 0; q < 12; ++q) *(v4h*)&smA[w][row][q * 4] = z4;
        }
    } else {                 // cols 48..95 (92..95 zero pad)
        if (valid) {
#pragma unroll
            for (int q = 12; q < 23; ++q)
                *(v4h*)&smA[w][row][q * 4] = __builtin_convertvector(xr[q], v4h);
        } else {
#pragma unroll
            for (int q = 12; q < 23; ++q) *(v4h*)&smA[w][row][q * 4] = z4;
        }
        *(v4h*)&smA[w][row][92] = z4;
    }
    if (t < 2 * D) smS[t] = 0.f;
    __syncthreads();
    v8f zero = {};
    v8f acc[NT_N];
#pragma unroll
    for (int nt = 0; nt < NT_N; ++nt) acc[nt] = zero;
#pragma unroll
    for (int ks = 0; ks < KS_N; ++ks) {
        union { v16h v; v8h h8[2]; } A;
        int kb = ks * 32 + ((lane < 16) ? 0 : 8);  // A-frag: K {kb..kb+7, kb+16..kb+23}
        A.h8[0] = *(const v8h*)&smA[w][row][kb];
        A.h8[1] = *(const v8h*)&smA[w][row][kb + 16];
#pragma unroll
        for (int nt = 0; nt < NT_N; ++nt) {
            v16h B = *(const v16h*)&w16[((ks * NT_N + nt) * 32 + lane) * 16];
            acc[nt] = wmma16(A.v, B, acc[nt]);
        }
    }
    int mofs = (lane < 16) ? 0 : 8;
#pragma unroll
    for (int nt = 0; nt < NT_N; ++nt) {
        int col = nt * 16 + (lane & 15);
        float s = 0.f, q = 0.f;
#pragma unroll
        for (int vi = 0; vi < 8; ++vi) {
            float z = acc[nt][vi];
            s += z; q += z * z;
            int nd = tile * 16 + mofs + vi;
            if (nd < n) hout[nd * D + col] = z;
        }
        atomicAdd(&smS[col], s);
        atomicAdd(&smS[D + col], q);
    }
    __syncthreads();
    if (t < 2 * D) atomicAdd(&stats[t], smS[t]);
}

// ---- fold BN (mean/var from sums) + gamma/beta into scale/shift ----
__global__ void k_finalize(const float* __restrict__ stats, const float* __restrict__ g,
                           const float* __restrict__ beta, float cnt, int ncols,
                           float* __restrict__ ss) {
    int c = blockIdx.x * blockDim.x + threadIdx.x;
    if (c < ncols) {
        float m   = stats[c] / cnt;
        float var = stats[ncols + c] / cnt - m * m;
        float sc  = g[c] * rsqrtf(var + sg::BNEPS);
        ss[c] = sc;
        ss[ncols + c] = beta[c] - m * sc;
    }
}

// ---- vectorized silu(BN(z)) in place; nvec = n*64/4 ----
__global__ void k_silu(float* __restrict__ h, int nvec, const float* __restrict__ ss) {
    int i = blockIdx.x * blockDim.x + threadIdx.x;
    if (i < nvec) {
        v4f x = ((const v4f*)h)[i];
        int col = (i * 4) & 63;
#pragma unroll
        for (int e = 0; e < 4; ++e) {
            float v = x[e] * ss[col + e] + ss[64 + col + e];
            x[e] = v * sgm(v);
        }
        ((v4f*)h)[i] = x;
    }
}

// ---- edge GEMM (WMMA). MODE 0: accumulate column stats. MODE 1: BN+gate*scr+scatter ----
template <int MODE>
__global__ void k_edge(const float* __restrict__ h, const float* __restrict__ ef,
                       const int* __restrict__ src, const int* __restrict__ dst,
                       const _Float16* __restrict__ w16, int nE,
                       float* __restrict__ stats, const float* __restrict__ ss,
                       float* __restrict__ upd) {
    using namespace sg;
    __shared__ _Float16 smA[4][16][KCP];
    __shared__ int smD[4][16];
    __shared__ float smS[256];
    int t = threadIdx.x, w = t >> 5, lane = t & 31;
    int tile = blockIdx.x * 4 + w;
    int row = lane & 15;
    int edge = tile * 16 + row;
    bool valid = edge < nE;
    int is = valid ? src[edge] : 0;
    int id = valid ? dst[edge] : 0;
    if (lane < 16) smD[w][row] = id;
    int seg = lane >> 4;
    v4h z4 = {};
    const v4f* hs4 = (const v4f*)(h + (size_t)is * D);   // 16B aligned rows
    const v4f* hd4 = (const v4f*)(h + (size_t)id * D);
    if (seg == 0) {          // comb cols 0..95  = hsrc[0..63] ++ hdst[0..31]
        if (valid) {
#pragma unroll
            for (int q = 0; q < 16; ++q)
                *(v4h*)&smA[w][row][q * 4] = __builtin_convertvector(hs4[q], v4h);
#pragma unroll
            for (int q = 0; q < 8; ++q)
                *(v4h*)&smA[w][row][64 + q * 4] = __builtin_convertvector(hd4[q], v4h);
        } else {
#pragma unroll
            for (int q = 0; q < 24; ++q) *(v4h*)&smA[w][row][q * 4] = z4;
        }
    } else {                 // comb cols 96..191 = hdst[32..63] ++ ef[0..40] ++ pad
#pragma unroll
        for (int q = 0; q < 6; ++q) *(v4h*)&smA[w][row][168 + q * 4] = z4;  // 168..191 (168 rewritten below)
        if (valid) {
#pragma unroll
            for (int q = 8; q < 16; ++q)
                *(v4h*)&smA[w][row][64 + q * 4] = __builtin_convertvector(hd4[q], v4h);
            const float* er = ef + (size_t)edge * EF;
#pragma unroll
            for (int c = 0; c < EF; ++c) smA[w][row][2 * D + c] = (_Float16)er[c];
        } else {
#pragma unroll
            for (int q = 0; q < 18; ++q) *(v4h*)&smA[w][row][96 + q * 4] = z4; // 96..167
        }
    }
    if (MODE == 0 && t < 128) { smS[t] = 0.f; smS[128 + t] = 0.f; }
    __syncthreads();
    v8f zero = {};
    v8f acc[NT_E];
#pragma unroll
    for (int nt = 0; nt < NT_E; ++nt) acc[nt] = zero;
#pragma unroll
    for (int ks = 0; ks < KS_E; ++ks) {
        union { v16h v; v8h h8[2]; } A;
        int kb = ks * 32 + ((lane < 16) ? 0 : 8);
        A.h8[0] = *(const v8h*)&smA[w][row][kb];
        A.h8[1] = *(const v8h*)&smA[w][row][kb + 16];
#pragma unroll
        for (int nt = 0; nt < NT_E; ++nt) {
            v16h B = *(const v16h*)&w16[((ks * NT_E + nt) * 32 + lane) * 16];
            acc[nt] = wmma16(A.v, B, acc[nt]);
        }
    }
    if (MODE == 0) {
#pragma unroll
        for (int nt = 0; nt < NT_E; ++nt) {
            int col = nt * 16 + (lane & 15);
            float s = 0.f, q = 0.f;
#pragma unroll
            for (int vi = 0; vi < 8; ++vi) { float z = acc[nt][vi]; s += z; q += z * z; }
            atomicAdd(&smS[col], s);
            atomicAdd(&smS[128 + col], q);
        }
        __syncthreads();
        if (t < 128) {
            atomicAdd(&stats[t], smS[t]);
            atomicAdd(&stats[128 + t], smS[128 + t]);
        }
    } else {
        int mofs = (lane < 16) ? 0 : 8;
#pragma unroll
        for (int nt = 0; nt < 4; ++nt) {
            int col = nt * 16 + (lane & 15);
            float scG = ss[col],      shG = ss[128 + col];
            float scS = ss[64 + col], shS = ss[192 + col];
#pragma unroll
            for (int vi = 0; vi < 8; ++vi) {
                int M = mofs + vi;
                int e2 = tile * 16 + M;
                if (e2 < nE) {
                    float gg = sgm(acc[nt][vi] * scG + shG);
                    float pp = splus(acc[nt + 4][vi] * scS + shS);
                    atomicAdd(&upd[smD[w][M] * D + col], gg * pp);
                }
            }
        }
    }
}

// ---- per-column stats over [n][64] array ----
__global__ void k_colstats64(const float* __restrict__ x, int n, float* __restrict__ stats) {
    __shared__ float sm[128];
    int t = threadIdx.x;
    if (t < 128) sm[t] = 0.f;
    __syncthreads();
    int col = t & 63;
    float s = 0.f, q = 0.f;
    for (int r = blockIdx.x * 2 + (t >> 6); r < n; r += gridDim.x * 2) {
        float v = x[r * 64 + col];
        s += v; q += v * v;
    }
    atomicAdd(&sm[col], s);
    atomicAdd(&sm[64 + col], q);
    __syncthreads();
    if (t < 128) atomicAdd(&stats[t], sm[t]);
}

// ---- vectorized h = softplus(BN(upd) + h); nvec = n*64/4 ----
__global__ void k_nodeup(float* __restrict__ h, const float* __restrict__ upd, int nvec,
                         const float* __restrict__ ss) {
    int i = blockIdx.x * blockDim.x + threadIdx.x;
    if (i < nvec) {
        v4f u = ((const v4f*)upd)[i];
        v4f x = ((const v4f*)h)[i];
        int col = (i * 4) & 63;
#pragma unroll
        for (int e = 0; e < 4; ++e)
            x[e] = splus(u[e] * ss[col + e] + ss[64 + col + e] + x[e]);
        ((v4f*)h)[i] = x;
    }
}

__global__ void k_pool(const float* __restrict__ h, const int* __restrict__ batch, int n,
                       float* __restrict__ psum, float* __restrict__ pcnt) {
    int i = blockIdx.x * blockDim.x + threadIdx.x;
    if (i < n * 64) {
        int node = i >> 6, col = i & 63;
        int g = batch[node];
        atomicAdd(&psum[g * 64 + col], h[i]);
        if (col == 0) atomicAdd(&pcnt[g], 1.f);
    }
}

// ---- head: concat pools -> FC(BN,silu) -> pred. One workgroup (NG=64 rows). ----
__global__ void k_head(const float* __restrict__ psS, const float* __restrict__ pcS,
                       const float* __restrict__ psB, const float* __restrict__ pcB,
                       const float* __restrict__ fcW, const float* __restrict__ fcG,
                       const float* __restrict__ fcBeta, const float* __restrict__ predW,
                       const float* __restrict__ predB, float* __restrict__ out) {
    using namespace sg;
    __shared__ float vt[NGn][FCn];
    int t = threadIdx.x;
    for (int i = t; i < NGn * FCn; i += blockDim.x) {
        int r = i >> 7, k = i & 127;
        float v = (k < 64) ? psS[r * 64 + k] / fmaxf(pcS[r], 1.f)
                           : psB[r * 64 + (k - 64)] / fmaxf(pcB[r], 1.f);
        vt[r][k] = v;
    }
    __syncthreads();
    int c = t;  // 128 threads = 128 FC columns
    float z[NGn];
    for (int r = 0; r < NGn; ++r) {
        float s = 0.f;
        for (int k = 0; k < FCn; ++k) s += vt[r][k] * fcW[k * FCn + c];
        z[r] = s;
    }
    float m = 0.f;
    for (int r = 0; r < NGn; ++r) m += z[r];
    m *= (1.f / NGn);
    float var = 0.f;
    for (int r = 0; r < NGn; ++r) { float d = z[r] - m; var += d * d; }
    var *= (1.f / NGn);
    float sc = fcG[c] * rsqrtf(var + BNEPS), sh = fcBeta[c] - m * sc;
    __syncthreads();  // everyone done reading vt
    for (int r = 0; r < NGn; ++r) {
        float xv = z[r] * sc + sh;
        vt[r][c] = xv * sgm(xv);
    }
    __syncthreads();
    if (t < NGn) {
        float s = predB[0];
        for (int k = 0; k < FCn; ++k) s += vt[t][k] * predW[k];
        out[t] = s;
    }
}

extern "C" void kernel_launch(void* const* d_in, const int* in_sizes, int n_in,
                              void* d_out, int out_size, void* d_ws, size_t ws_size,
                              hipStream_t stream) {
    using namespace sg;
    (void)in_sizes; (void)n_in; (void)out_size; (void)ws_size;
    const float* vs_feat = (const float*)d_in[0];
    const float* es_feat = (const float*)d_in[1];
    const float* vb_feat = (const float*)d_in[2];
    const float* eb_feat = (const float*)d_in[3];
    const int* src_s = (const int*)d_in[4];
    const int* dst_s = (const int*)d_in[5];
    const int* batch_s = (const int*)d_in[6];
    const int* src_b = (const int*)d_in[7];
    const int* dst_b = (const int*)d_in[8];
    const int* batch_b = (const int*)d_in[9];
    const float* emb_s_W = (const float*)d_in[10];
    const float* emb_s_g = (const float*)d_in[12];
    const float* emb_s_beta = (const float*)d_in[13];
    const float* emb_b_W = (const float*)d_in[14];
    const float* emb_b_g = (const float*)d_in[16];
    const float* emb_b_beta = (const float*)d_in[17];
    const float* conv_s_W = (const float*)d_in[18];
    const float* conv_s_g = (const float*)d_in[20];
    const float* conv_s_beta = (const float*)d_in[21];
    const float* conv_s_ng = (const float*)d_in[22];
    const float* conv_s_nb = (const float*)d_in[23];
    const float* conv_b_W = (const float*)d_in[24];
    const float* conv_b_g = (const float*)d_in[26];
    const float* conv_b_beta = (const float*)d_in[27];
    const float* conv_b_ng = (const float*)d_in[28];
    const float* conv_b_nb = (const float*)d_in[29];
    const float* fc_W = (const float*)d_in[30];
    const float* fc_g = (const float*)d_in[32];
    const float* fc_beta = (const float*)d_in[33];
    const float* pred_W = (const float*)d_in[34];
    const float* pred_b = (const float*)d_in[35];

    char* ws = (char*)d_ws;
    size_t off = 0;
    _Float16* w16 = (_Float16*)(ws + off);
    const int W16_HALVES = 2 * EMB_H + 2 * LY * CONV_H;  // 159744
    off += (size_t)W16_HALVES * 2;
    off = (off + 511) & ~(size_t)511;
    float* h_s  = (float*)(ws + off); off += (size_t)NSn * D * 4;
    float* h_b  = (float*)(ws + off); off += (size_t)NBn * D * 4;
    float* updv = (float*)(ws + off); off += (size_t)NSn * D * 4;
    float* SR   = (float*)(ws + off);
    float* stats_emb[2] = { SR, SR + 128 };
    float* stats_edge = SR + 256;           // 6 x 256
    float* stats_node = SR + 1792;          // 6 x 128
    float* ss_emb[2] = { SR + 2560, SR + 2688 };
    float* ss_edge = SR + 2816;             // 6 x 256
    float* ss_node = SR + 4352;             // 6 x 128
    float* psum[2] = { SR + 5120, SR + 9280 };
    float* pcnt[2] = { SR + 9216, SR + 13376 };
    const int SR_FLOATS = 13440;

    k_prep<<<(W16_HALVES + 255) / 256, 256, 0, stream>>>(emb_s_W, emb_b_W, conv_s_W, conv_b_W, w16);
    k_fill<<<(SR_FLOATS + 255) / 256, 256, 0, stream>>>(SR, SR_FLOATS);

    for (int br = 0; br < 2; ++br) {
        const float* vf  = br ? vb_feat : vs_feat;
        const float* efe = br ? eb_feat : es_feat;
        const int* srcp = br ? src_b : src_s;
        const int* dstp = br ? dst_b : dst_s;
        const int* bat  = br ? batch_b : batch_s;
        float* hbuf = br ? h_b : h_s;
        int nN = br ? NBn : NSn;
        int nE = br ? EBn : ESn;
        const _Float16* w16emb  = w16 + br * EMB_H;
        const _Float16* w16conv = w16 + 2 * EMB_H + br * (LY * CONV_H);
        const float* embg    = br ? emb_b_g : emb_s_g;
        const float* embbeta = br ? emb_b_beta : emb_s_beta;
        const float* cg    = br ? conv_b_g : conv_s_g;
        const float* cbeta = br ? conv_b_beta : conv_s_beta;
        const float* cng   = br ? conv_b_ng : conv_s_ng;
        const float* cnb   = br ? conv_b_nb : conv_s_nb;

        int tilesN = (nN + 15) / 16, gN = (tilesN + 3) / 4;
        k_emb<<<gN, 128, 0, stream>>>(vf, nN, w16emb, hbuf, stats_emb[br]);
        k_finalize<<<1, 128, 0, stream>>>(stats_emb[br], embg, embbeta, (float)nN, 64, ss_emb[br]);
        k_silu<<<(nN * 16 + 255) / 256, 256, 0, stream>>>(hbuf, nN * 16, ss_emb[br]);

        for (int l = 0; l < LY; ++l) {
            float* stE = stats_edge + (br * 3 + l) * 256;
            float* stN = stats_node + (br * 3 + l) * 128;
            float* sE  = ss_edge + (br * 3 + l) * 256;
            float* sN  = ss_node + (br * 3 + l) * 128;
            const _Float16* wl = w16conv + l * CONV_H;
            k_fill<<<(nN * 64 + 255) / 256, 256, 0, stream>>>(updv, nN * 64);
            int tilesE = (nE + 15) / 16, gE = (tilesE + 3) / 4;
            k_edge<0><<<gE, 128, 0, stream>>>(hbuf, efe, srcp, dstp, wl, nE, stE, nullptr, nullptr);
            k_finalize<<<1, 128, 0, stream>>>(stE, cg + l * 128, cbeta + l * 128, (float)nE, 128, sE);
            k_edge<1><<<gE, 128, 0, stream>>>(hbuf, efe, srcp, dstp, wl, nE, nullptr, sE, updv);
            k_colstats64<<<512, 128, 0, stream>>>(updv, nN, stN);
            k_finalize<<<1, 128, 0, stream>>>(stN, cng + l * 64, cnb + l * 64, (float)nN, 64, sN);
            k_nodeup<<<(nN * 16 + 255) / 256, 256, 0, stream>>>(hbuf, updv, nN * 16, sN);
        }
        k_pool<<<(nN * 64 + 255) / 256, 256, 0, stream>>>(hbuf, bat, nN, psum[br], pcnt[br]);
    }
    k_head<<<1, 128, 0, stream>>>(psum[0], pcnt[0], psum[1], pcnt[1],
                                  fc_W, fc_g, fc_beta, pred_W, pred_b, (float*)d_out);
}
